// MaterialsGraphSAGE_14061722927385
// MI455X (gfx1250) — matrix operand order repaired and checked by reading, not compiled, not run on hardware
//
#include <hip/hip_runtime.h>
#include <hip/hip_bf16.h>

#define N_NODES 50000
#define N_EDGES 600000
#define N_GRAPHS 256
#define NODE_F 64
#define GLOB_F 16
#define HDIM 128
#define NLAYERS 4
#define NEG_SLOPE 0.01f
#define INV_STD 0.9999950000374997f   // 1/sqrt(1 + 1e-5)

typedef __attribute__((ext_vector_type(16))) _Float16 v16h;
typedef __attribute__((ext_vector_type(8)))  float    v8f;

__device__ __forceinline__ float lrelu(float v) {
    return v >= 0.0f ? v : NEG_SLOPE * v;
}

// ---------------------------------------------------------------------------
// A fragment (16x32 f16) from an f32 row-major matrix, rows [row0,row0+16),
// cols [k0,k0+32).  ISA layout: lane L row = L&15;
// element e -> K = (e&7) + 8*(L>>4) + 16*(e>>3)  =>  per lane this is two
// contiguous 8-float runs at rp[0..7] and rp[16..23] -> 4x global_load_b128.
// ---------------------------------------------------------------------------
__device__ __forceinline__ v16h load_a_f32(const float* __restrict__ A, int ld,
                                           int row0, int k0, int lane, float scale) {
    const float* rp = A + (size_t)(row0 + (lane & 15)) * ld + k0 + 8 * (lane >> 4);
    const float4 f0 = *(const float4*)(rp + 0);
    const float4 f1 = *(const float4*)(rp + 4);
    const float4 f2 = *(const float4*)(rp + 16);
    const float4 f3 = *(const float4*)(rp + 20);
    v16h a;
    a[0]  = (_Float16)(f0.x * scale); a[1]  = (_Float16)(f0.y * scale);
    a[2]  = (_Float16)(f0.z * scale); a[3]  = (_Float16)(f0.w * scale);
    a[4]  = (_Float16)(f1.x * scale); a[5]  = (_Float16)(f1.y * scale);
    a[6]  = (_Float16)(f1.z * scale); a[7]  = (_Float16)(f1.w * scale);
    a[8]  = (_Float16)(f2.x * scale); a[9]  = (_Float16)(f2.y * scale);
    a[10] = (_Float16)(f2.z * scale); a[11] = (_Float16)(f2.w * scale);
    a[12] = (_Float16)(f3.x * scale); a[13] = (_Float16)(f3.y * scale);
    a[14] = (_Float16)(f3.z * scale); a[15] = (_Float16)(f3.w * scale);
    return a;
}

// ---------------------------------------------------------------------------
// Pre-swizzled B fragments in LDS: layout [frag][lane][16 halves], where
// frag = kt*8 + j.  Each lane reads its 32 contiguous bytes -> 2x ds_load_b128.
// Swizzle fill mapping (B 32x16 fragment): lane L col = L&15,
// element e -> K = kt*32 + (e&7) + 8*(L>>4) + 16*(e>>3).
// ---------------------------------------------------------------------------
__device__ __forceinline__ v16h load_b_frag(const _Float16* __restrict__ sW,
                                            int kt, int j, int lane) {
    return *(const v16h*)(sW + (size_t)(((kt << 3) + j) * 32 + lane) * 16);
}

__device__ __forceinline__ void swizzle_fill(const float* __restrict__ W,
                                             _Float16* __restrict__ sW,
                                             int total /* K*HDIM */) {
    for (int idx = threadIdx.x; idx < total; idx += blockDim.x) {
        const int e    = idx & 15;
        const int lane = (idx >> 4) & 31;
        const int f    = idx >> 9;          // frag index
        const int kt   = f >> 3;
        const int j    = f & 7;
        const int k    = kt * 32 + (e & 7) + 8 * (lane >> 4) + 16 * (e >> 3);
        const int n    = j * 16 + (lane & 15);
        sW[idx] = (_Float16)W[(size_t)k * HDIM + n];
    }
}

// ---------------- embedding: h = lrelu(x @ W_emb + b_emb) ----------------
// Weight staged global->LDS with gfx1250 async-to-LDS ops (ASYNCcnt), then
// swizzled to f16 WMMA fragment layout.
__global__ void gs_embed_kernel(const float* __restrict__ x,
                                const float* __restrict__ Wemb,
                                const float* __restrict__ bemb,
                                float* __restrict__ h) {
    __shared__ __align__(16) float    sStage[NODE_F * HDIM];  // 32 KB raw f32
    __shared__ __align__(32) _Float16 sW[NODE_F * HDIM];      // 16 KB swizzled f16

    // async copy Wemb (8192 floats) into LDS: one b128 per lane per iter
    {
        const unsigned ldsBase = (unsigned)(size_t)(&sStage[0]); // LDS aperture low bits
        for (int i = threadIdx.x; i < (NODE_F * HDIM) / 4; i += blockDim.x) {
            const unsigned laddr = ldsBase + (unsigned)i * 16u;
            const unsigned long long gaddr = (unsigned long long)(Wemb + (size_t)i * 4);
            asm volatile("global_load_async_to_lds_b128 %0, %1, off"
                         :: "v"(laddr), "v"(gaddr) : "memory");
        }
        asm volatile("s_wait_asynccnt 0x0" ::: "memory");
    }
    __syncthreads();

    // swizzle f32 staging -> f16 fragment layout
    for (int idx = threadIdx.x; idx < NODE_F * HDIM; idx += blockDim.x) {
        const int e    = idx & 15;
        const int lane = (idx >> 4) & 31;
        const int f    = idx >> 9;
        const int kt   = f >> 3;
        const int j    = f & 7;
        const int k    = kt * 32 + (e & 7) + 8 * (lane >> 4) + 16 * (e >> 3);
        const int n    = j * 16 + (lane & 15);
        sW[idx] = (_Float16)sStage[k * HDIM + n];
    }
    __syncthreads();

    const int lane = threadIdx.x & 31;
    const int tile = blockIdx.x * (blockDim.x >> 5) + (threadIdx.x >> 5);
    if (tile >= N_NODES / 16) return;
    const int row0 = tile * 16;

    v8f acc[8];
#pragma unroll
    for (int j = 0; j < 8; ++j)
#pragma unroll
        for (int r = 0; r < 8; ++r) acc[j][r] = 0.0f;

#pragma unroll
    for (int kt = 0; kt < NODE_F / 32; ++kt) {
        v16h a = load_a_f32(x, NODE_F, row0, kt * 32, lane, 1.0f);
#pragma unroll
        for (int j = 0; j < 8; ++j) {
            v16h b = load_b_frag(sW, kt, j, lane);
            acc[j] = __builtin_amdgcn_wmma_f32_16x16x32_f16(
                false, a, false, b, (short)0, acc[j], false, false);
        }
    }

    const int nlo = lane & 15;
    const int mhi = 8 * (lane >> 4);
#pragma unroll
    for (int j = 0; j < 8; ++j) {
        const int n = j * 16 + nlo;
        const float bb = bemb[n];
#pragma unroll
        for (int r = 0; r < 8; ++r) {
            const int row = row0 + r + mhi;
            h[(size_t)row * HDIM + n] = lrelu(acc[j][r] + bb);
        }
    }
}

// ---------------- per-destination degree (computed once) ----------------
__global__ void gs_degree_kernel(const int* __restrict__ dst, float* __restrict__ cnt) {
    const int e = blockIdx.x * blockDim.x + threadIdx.x;
    if (e < N_EDGES) atomicAdd(&cnt[dst[e]], 1.0f);
}

// ---------------- edge scatter: summed[dst] += h[src] ----------------
__global__ void gs_scatter_kernel(const float* __restrict__ h,
                                  const int* __restrict__ src,
                                  const int* __restrict__ dst,
                                  float* __restrict__ summed) {
    const size_t idx = (size_t)blockIdx.x * blockDim.x + threadIdx.x;
    if (idx >= (size_t)N_EDGES * HDIM) return;
    const int e = (int)(idx >> 7);       // / HDIM
    const int f = (int)(idx & (HDIM - 1));
    const int s = src[e];
    const int d = dst[e];
    atomicAdd(&summed[(size_t)d * HDIM + f], h[(size_t)s * HDIM + f]);
}

// ---- SAGE layer: h' = BN(lrelu(mean @ W_l + b_l + h @ W_r)) fused, K=256 WMMA ----
__global__ void gs_sage_kernel(const float* __restrict__ h,
                               const float* __restrict__ summed,
                               const float* __restrict__ cnt,
                               const float* __restrict__ Wl,
                               const float* __restrict__ bl,
                               const float* __restrict__ Wr,
                               const float* __restrict__ gamma,
                               const float* __restrict__ beta,
                               float* __restrict__ hout) {
    // W_r frags then W_l frags, pre-swizzled, exactly 64 KB
    __shared__ __align__(32) _Float16 sW[2 * HDIM * HDIM];
    swizzle_fill(Wr, sW, HDIM * HDIM);
    swizzle_fill(Wl, sW + HDIM * HDIM, HDIM * HDIM);
    __syncthreads();

    const int lane = threadIdx.x & 31;
    const int tile = blockIdx.x * (blockDim.x >> 5) + (threadIdx.x >> 5);
    if (tile >= N_NODES / 16) return;
    const int row0 = tile * 16;

    // per-lane reciprocal degree for this lane's A row
    const float rinv = 1.0f / fmaxf(cnt[row0 + (lane & 15)], 1.0f);

    v8f acc[8];
#pragma unroll
    for (int j = 0; j < 8; ++j)
#pragma unroll
        for (int r = 0; r < 8; ++r) acc[j][r] = 0.0f;

    // root half: h @ W_r
#pragma unroll
    for (int kt = 0; kt < HDIM / 32; ++kt) {
        v16h a = load_a_f32(h, HDIM, row0, kt * 32, lane, 1.0f);
#pragma unroll
        for (int j = 0; j < 8; ++j) {
            v16h b = load_b_frag(sW, kt, j, lane);
            acc[j] = __builtin_amdgcn_wmma_f32_16x16x32_f16(
                false, a, false, b, (short)0, acc[j], false, false);
        }
    }
    // neighbor half: (summed/deg) @ W_l
#pragma unroll
    for (int kt = 0; kt < HDIM / 32; ++kt) {
        v16h a = load_a_f32(summed, HDIM, row0, kt * 32, lane, rinv);
#pragma unroll
        for (int j = 0; j < 8; ++j) {
            v16h b = load_b_frag(sW + HDIM * HDIM, kt, j, lane);
            acc[j] = __builtin_amdgcn_wmma_f32_16x16x32_f16(
                false, a, false, b, (short)0, acc[j], false, false);
        }
    }

    const int nlo = lane & 15;
    const int mhi = 8 * (lane >> 4);
#pragma unroll
    for (int j = 0; j < 8; ++j) {
        const int n = j * 16 + nlo;
        const float bb = bl[n];
        const float g  = gamma[n] * INV_STD;
        const float be = beta[n];
#pragma unroll
        for (int r = 0; r < 8; ++r) {
            const int row = row0 + r + mhi;
            float v = lrelu(acc[j][r] + bb);
            hout[(size_t)row * HDIM + n] = v * g + be;
        }
    }
}

// ---------------- global mean pool (sum + counts via atomics) ----------------
__global__ void gs_pool_feat_kernel(const float* __restrict__ h,
                                    const int* __restrict__ batch,
                                    float* __restrict__ pooled) {
    const size_t idx = (size_t)blockIdx.x * blockDim.x + threadIdx.x;
    if (idx >= (size_t)N_NODES * HDIM) return;
    const int node = (int)(idx >> 7);
    const int f    = (int)(idx & (HDIM - 1));
    const int g    = batch[node];
    atomicAdd(&pooled[(size_t)g * HDIM + f], h[(size_t)node * HDIM + f]);
}

__global__ void gs_pool_cnt_kernel(const int* __restrict__ batch, float* __restrict__ gcnt) {
    const int node = blockIdx.x * blockDim.x + threadIdx.x;
    if (node < N_NODES) atomicAdd(&gcnt[batch[node]], 1.0f);
}

// ---------------- head MLP: one block (128 threads) per graph ----------------
__global__ void gs_head_kernel(const float* __restrict__ pooled,
                               const float* __restrict__ gcnt,
                               const float* __restrict__ u,
                               const float* __restrict__ Wg,
                               const float* __restrict__ bg,
                               const float* __restrict__ Wf1,
                               const float* __restrict__ bf1,
                               const float* __restrict__ Wf2,
                               const float* __restrict__ bf2,
                               float* __restrict__ out) {
    __shared__ float pm[HDIM];
    __shared__ float ugs[HDIM];
    __shared__ float hid[HDIM];
    const int g = blockIdx.x;
    const int n = threadIdx.x;

    const float inv = 1.0f / fmaxf(gcnt[g], 1.0f);
    pm[n] = pooled[(size_t)g * HDIM + n] * inv;

    float a = bg[n];
#pragma unroll
    for (int k = 0; k < GLOB_F; ++k) a += u[g * GLOB_F + k] * Wg[k * HDIM + n];
    ugs[n] = lrelu(a);
    __syncthreads();

    float hsum = bf1[n];
    for (int k = 0; k < HDIM; ++k) hsum += pm[k]  * Wf1[(size_t)k * HDIM + n];
    for (int k = 0; k < HDIM; ++k) hsum += ugs[k] * Wf1[(size_t)(HDIM + k) * HDIM + n];
    hid[n] = lrelu(hsum);
    __syncthreads();

    if (n < 2) {
        float o = bf2[n];
        for (int k = 0; k < HDIM; ++k) o += hid[k] * Wf2[k * 2 + n];
        out[g * 2 + n] = o;
    }
}

extern "C" void kernel_launch(void* const* d_in, const int* in_sizes, int n_in,
                              void* d_out, int out_size, void* d_ws, size_t ws_size,
                              hipStream_t stream) {
    (void)in_sizes; (void)n_in; (void)out_size; (void)ws_size;

    const float* x      = (const float*)d_in[0];
    const int*   ei     = (const int*)  d_in[1];
    const float* u      = (const float*)d_in[2];
    const int*   batch  = (const int*)  d_in[3];
    const float* W_emb  = (const float*)d_in[4];
    const float* b_emb  = (const float*)d_in[5];
    const float* W_l    = (const float*)d_in[6];
    const float* b_l    = (const float*)d_in[7];
    const float* W_r    = (const float*)d_in[8];
    const float* gamma  = (const float*)d_in[9];
    const float* beta   = (const float*)d_in[10];
    const float* W_g    = (const float*)d_in[11];
    const float* b_g    = (const float*)d_in[12];
    const float* W_f1   = (const float*)d_in[13];
    const float* b_f1   = (const float*)d_in[14];
    const float* W_f2   = (const float*)d_in[15];
    const float* b_f2   = (const float*)d_in[16];
    float* out = (float*)d_out;

    const int* src = ei;
    const int* dst = ei + N_EDGES;

    // workspace layout (floats)
    const size_t NH = (size_t)N_NODES * HDIM;
    float* h0     = (float*)d_ws;
    float* h1     = h0 + NH;
    float* summed = h1 + NH;
    float* cnt    = summed + NH;
    float* pooled = cnt + N_NODES;
    float* gcnt   = pooled + (size_t)N_GRAPHS * HDIM;

    const int ROW_TILES = N_NODES / 16;                 // 3125
    const int WPB = 8;                                  // waves per block
    const int gemm_blocks = (ROW_TILES + WPB - 1) / WPB;

    // 1. embedding
    gs_embed_kernel<<<gemm_blocks, WPB * 32, 0, stream>>>(x, W_emb, b_emb, h0);

    // 2. degrees (reused by all layers)
    hipMemsetAsync(cnt, 0, N_NODES * sizeof(float), stream);
    gs_degree_kernel<<<(N_EDGES + 255) / 256, 256, 0, stream>>>(dst, cnt);

    // 3. SAGE layers
    float* hcur = h0;
    float* hnext = h1;
    const int scatter_blocks = (int)(((size_t)N_EDGES * HDIM + 255) / 256);
    for (int i = 0; i < NLAYERS; ++i) {
        hipMemsetAsync(summed, 0, NH * sizeof(float), stream);
        gs_scatter_kernel<<<scatter_blocks, 256, 0, stream>>>(hcur, src, dst, summed);
        gs_sage_kernel<<<gemm_blocks, WPB * 32, 0, stream>>>(
            hcur, summed, cnt,
            W_l + (size_t)i * HDIM * HDIM, b_l + (size_t)i * HDIM,
            W_r + (size_t)i * HDIM * HDIM,
            gamma + (size_t)i * HDIM, beta + (size_t)i * HDIM, hnext);
        float* t = hcur; hcur = hnext; hnext = t;
    }

    // 4. pooling
    hipMemsetAsync(pooled, 0, (size_t)N_GRAPHS * HDIM * sizeof(float), stream);
    hipMemsetAsync(gcnt, 0, N_GRAPHS * sizeof(float), stream);
    gs_pool_feat_kernel<<<(int)((NH + 255) / 256), 256, 0, stream>>>(hcur, batch, pooled);
    gs_pool_cnt_kernel<<<(N_NODES + 255) / 256, 256, 0, stream>>>(batch, gcnt);

    // 5. head MLP
    gs_head_kernel<<<N_GRAPHS, HDIM, 0, stream>>>(
        pooled, gcnt, u, W_g, b_g, W_f1, b_f1, W_f2, b_f2, out);
}
